// GATWithMLPLinkPred_83459804496443
// MI455X (gfx1250) — compile-verified, matrix operand order checked by reading, not run on hardware
//
#include <hip/hip_runtime.h>
#include <math.h>

typedef __attribute__((ext_vector_type(16))) __bf16 v16bf;
typedef __attribute__((ext_vector_type(8)))  float  v8f;

#define N_NODES 50000
#define E_EDGES 500000
#define E_TOT   (E_EDGES + N_NODES)   // self-loops appended
#define IN_CH   128
#define HID     32
#define HEADS   8
#define F1      (HEADS * HID)         // 256
#define OUT_CH  64

// ---------------------------------------------------------------------------
// helpers
// ---------------------------------------------------------------------------
__device__ __forceinline__ float atomic_max_f32(float* addr, float val) {
  // float-ordered atomic max via integer atomics (sign-split trick)
  if (val >= 0.0f)
    return __int_as_float(atomicMax((int*)addr, __float_as_int(val)));
  return __uint_as_float(atomicMin((unsigned int*)addr, __float_as_uint(val)));
}

__device__ __forceinline__ int edge_src(const int* __restrict__ ei, int e) {
  return (e < E_EDGES) ? ei[e] : (e - E_EDGES);
}
__device__ __forceinline__ int edge_dst(const int* __restrict__ ei, int e) {
  return (e < E_EDGES) ? ei[E_EDGES + e] : (e - E_EDGES);
}

__global__ void fill_f32(float* __restrict__ p, float v, int n) {
  int i = blockIdx.x * blockDim.x + threadIdx.x;
  if (i < n) p[i] = v;
}

// ---------------------------------------------------------------------------
// WMMA GEMM:  Hout[M x Nc] = X[M x K] * W^T,  W is [Nc x K] row-major.
// One wave (32 lanes) computes one 16x16 output tile using
// v_wmma_f32_16x16x32_bf16 (fp32 loads converted to bf16, f32 accumulate).
// M must be a multiple of 16 (50000 = 3125*16), Nc mult of 16, K mult of 32.
// ---------------------------------------------------------------------------
template <int K>
__global__ void gemm_xwt_wmma(const float* __restrict__ X,
                              const float* __restrict__ W,
                              float* __restrict__ Hout,
                              int M, int Nc) {
  const int wave  = (blockIdx.x * blockDim.x + threadIdx.x) >> 5;
  const int lane  = threadIdx.x & 31;
  const int ntile = Nc >> 4;
  const int row0  = (wave / ntile) << 4;
  const int col0  = (wave % ntile) << 4;
  if (row0 >= M) return;

  const int m    = lane & 15;   // A-row for this lane (also B-col index n)
  const int half = lane >> 4;   // half-wave select

  v8f acc = {};
  const float* xrow = X + (size_t)(row0 + m) * K;   // A row m
  const float* wrow = W + (size_t)(col0 + m) * K;   // B col n == m

  for (int k0 = 0; k0 < K; k0 += 32) {
    v16bf a, b;
#pragma unroll
    for (int v = 0; v < 8; ++v) {
      // ISA 16-bit 16x32 A layout: VGPR v holds K = (v>>2)*16 + half*8 + (v&3)*2 (+1)
      const int kb = ((v >> 2) << 4) + (half << 3) + ((v & 3) << 1);
      float2 xa = *(const float2*)(xrow + k0 + kb);
      float2 wb = *(const float2*)(wrow + k0 + kb);
      a[2 * v]     = (__bf16)xa.x;
      a[2 * v + 1] = (__bf16)xa.y;
      b[2 * v]     = (__bf16)wb.x;
      b[2 * v + 1] = (__bf16)wb.y;
    }
    acc = __builtin_amdgcn_wmma_f32_16x16x32_bf16(
        /*neg_a=*/false, a, /*neg_b=*/false, b,
        /*c_mod=*/(short)0, acc, /*reuse_a=*/false, /*reuse_b=*/false);
  }

  // D layout: VGPR r -> row = half*8 + r, col = m
#pragma unroll
  for (int r = 0; r < 8; ++r)
    Hout[(size_t)(row0 + (half << 3) + r) * Nc + (col0 + m)] = acc[r];
}

// ---------------------------------------------------------------------------
// alpha_src / alpha_dst per (node, head):  dot over C channels
// ---------------------------------------------------------------------------
template <int H, int C>
__global__ void attn_coef(const float* __restrict__ Hf,
                          const float* __restrict__ a_src,
                          const float* __restrict__ a_dst,
                          float* __restrict__ as, float* __restrict__ ad) {
  int i = blockIdx.x * blockDim.x + threadIdx.x;   // i = n*H + h
  if (i >= N_NODES * H) return;
  int h = i % H;
  const float* hp = Hf + (size_t)i * C;
  const float* sp = a_src + h * C;
  const float* dp = a_dst + h * C;
  float s = 0.f, d = 0.f;
#pragma unroll
  for (int c = 0; c < C; c += 4) {
    float4 v  = *(const float4*)(hp + c);
    float4 va = *(const float4*)(sp + c);
    float4 vb = *(const float4*)(dp + c);
    s += v.x * va.x + v.y * va.y + v.z * va.z + v.w * va.w;
    d += v.x * vb.x + v.y * vb.y + v.z * vb.z + v.w * vb.w;
  }
  as[i] = s;
  ad[i] = d;
}

// ---------------------------------------------------------------------------
// pass 1: e = leaky_relu(as[src]+ad[dst]); segment max over dst (atomic)
// ---------------------------------------------------------------------------
template <int H>
__global__ void edge_lrelu_max(const int* __restrict__ ei,
                               const float* __restrict__ as,
                               const float* __restrict__ ad,
                               float* __restrict__ mbuf) {
  int i = blockIdx.x * blockDim.x + threadIdx.x;   // (edge, head)
  if (i >= E_TOT * H) return;
  int e = i / H, h = i - e * H;
  int s = edge_src(ei, e), d = edge_dst(ei, e);
  float v = as[s * H + h] + ad[d * H + h];
  v = v > 0.f ? v : 0.2f * v;
  atomic_max_f32(&mbuf[d * H + h], v);
}

// ---------------------------------------------------------------------------
// pass 2: ex = exp(e - m[dst]); store ex; segment sum denom (atomic)
// ---------------------------------------------------------------------------
template <int H>
__global__ void edge_exp_sum(const int* __restrict__ ei,
                             const float* __restrict__ as,
                             const float* __restrict__ ad,
                             const float* __restrict__ mbuf,
                             float* __restrict__ exbuf,
                             float* __restrict__ denom) {
  int i = blockIdx.x * blockDim.x + threadIdx.x;
  if (i >= E_TOT * H) return;
  int e = i / H, h = i - e * H;
  int s = edge_src(ei, e), d = edge_dst(ei, e);
  float v = as[s * H + h] + ad[d * H + h];
  v = v > 0.f ? v : 0.2f * v;
  float ex = __expf(v - mbuf[d * H + h]);
  exbuf[i] = ex;
  atomicAdd(&denom[d * H + h], ex);
}

// ---------------------------------------------------------------------------
// pass 2.5: normalize weights in place: w = ex / (denom[dst] + eps)
// (moves the denom gather + divide out of the 8x/16x-larger aggregation pass)
// ---------------------------------------------------------------------------
template <int H>
__global__ void edge_normalize(const int* __restrict__ ei,
                               const float* __restrict__ denom,
                               float* __restrict__ exbuf) {
  int i = blockIdx.x * blockDim.x + threadIdx.x;
  if (i >= E_TOT * H) return;
  int e = i / H, h = i - e * H;
  int d = edge_dst(ei, e);
  exbuf[i] = exbuf[i] / (denom[d * H + h] + 1e-16f);
}

// ---------------------------------------------------------------------------
// pass 3: agg[dst] += w * h[src]   (thread per (edge, 4 features))
// b128 gather of h[src]; 4 atomic f32 adds (all L2-resident)
// ---------------------------------------------------------------------------
template <int H, int C>
__global__ void edge_aggregate(const int* __restrict__ ei,
                               const float* __restrict__ wbuf,
                               const float* __restrict__ Hf,
                               float* __restrict__ agg) {
  constexpr int F4 = (H * C) / 4;
  int i = blockIdx.x * blockDim.x + threadIdx.x;   // (edge, f/4)
  if (i >= E_TOT * F4) return;
  int e  = i / F4;
  int f  = (i - e * F4) * 4;       // 4-aligned, never crosses a head (C % 4 == 0)
  int h  = f / C;
  int s = edge_src(ei, e), d = edge_dst(ei, e);
  float  w  = wbuf[e * H + h];
  float4 hv = *(const float4*)(Hf + (size_t)s * (H * C) + f);
  float* ap = agg + (size_t)d * (H * C) + f;
  atomicAdd(ap + 0, w * hv.x);
  atomicAdd(ap + 1, w * hv.y);
  atomicAdd(ap + 2, w * hv.z);
  atomicAdd(ap + 3, w * hv.w);
}

// ---------------------------------------------------------------------------
// z = elu(agg + bias1)   (in place)
// ---------------------------------------------------------------------------
__global__ void elu_bias1(float* __restrict__ z, const float* __restrict__ bias,
                          int total) {
  int i = blockIdx.x * blockDim.x + threadIdx.x;
  if (i >= total) return;
  float v = z[i] + bias[i % F1];
  z[i] = v > 0.f ? v : (__expf(v) - 1.f);
}

// out = agg2 + bias2
__global__ void add_bias_out(const float* __restrict__ agg,
                             const float* __restrict__ bias,
                             float* __restrict__ out, int total) {
  int i = blockIdx.x * blockDim.x + threadIdx.x;
  if (i >= total) return;
  out[i] = agg[i] + bias[i % OUT_CH];
}

// ---------------------------------------------------------------------------
extern "C" void kernel_launch(void* const* d_in, const int* in_sizes, int n_in,
                              void* d_out, int out_size, void* d_ws, size_t ws_size,
                              hipStream_t stream) {
  const float* x        = (const float*)d_in[0];
  const int*   ei       = (const int*)d_in[1];          // [2, E] int32
  const float* W1       = (const float*)d_in[2];        // [256,128]
  const float* att_src1 = (const float*)d_in[3];        // [8,32]
  const float* att_dst1 = (const float*)d_in[4];
  const float* bias1    = (const float*)d_in[5];        // [256]
  const float* W2       = (const float*)d_in[6];        // [64,256]
  const float* att_src2 = (const float*)d_in[7];        // [1,64]
  const float* att_dst2 = (const float*)d_in[8];
  const float* bias2    = (const float*)d_in[9];        // [64]
  float* out            = (float*)d_out;

  // ---- workspace layout (floats) ----
  float* Wp = (float*)d_ws;
  size_t o = 0;
  float* h1   = Wp + o; o += (size_t)N_NODES * F1;      // 12.8M
  float* aggz = Wp + o; o += (size_t)N_NODES * F1;      // 12.8M (agg1 -> z in place)
  float* as1  = Wp + o; o += (size_t)N_NODES * HEADS;
  float* ad1  = Wp + o; o += (size_t)N_NODES * HEADS;
  float* m1   = Wp + o; o += (size_t)N_NODES * HEADS;
  float* den1 = Wp + o; o += (size_t)N_NODES * HEADS;
  float* ex1  = Wp + o; o += (size_t)E_TOT * HEADS;     // 4.4M
  // layer-2 buffers overlay dead layer-1 storage
  float* h2   = h1;                                     // N*64
  float* agg2 = h1 + (size_t)N_NODES * OUT_CH;          // N*64
  float* as2  = as1;
  float* ad2  = ad1;
  float* m2   = m1;
  float* den2 = den1;
  float* ex2  = ex1;

  const dim3 B(256);
  auto blocks = [](long long t) { return dim3((unsigned)((t + 255) / 256)); };

  // =================== layer 1 ===================
  {
    long long waves = (long long)(N_NODES / 16) * (F1 / 16);
    gemm_xwt_wmma<IN_CH><<<blocks(waves * 32), B, 0, stream>>>(x, W1, h1, N_NODES, F1);
  }
  attn_coef<HEADS, HID><<<blocks((long long)N_NODES * HEADS), B, 0, stream>>>(
      h1, att_src1, att_dst1, as1, ad1);
  fill_f32<<<blocks((long long)N_NODES * HEADS), B, 0, stream>>>(
      m1, -INFINITY, N_NODES * HEADS);
  hipMemsetAsync(den1, 0, (size_t)N_NODES * HEADS * sizeof(float), stream);
  hipMemsetAsync(aggz, 0, (size_t)N_NODES * F1 * sizeof(float), stream);
  edge_lrelu_max<HEADS><<<blocks((long long)E_TOT * HEADS), B, 0, stream>>>(
      ei, as1, ad1, m1);
  edge_exp_sum<HEADS><<<blocks((long long)E_TOT * HEADS), B, 0, stream>>>(
      ei, as1, ad1, m1, ex1, den1);
  edge_normalize<HEADS><<<blocks((long long)E_TOT * HEADS), B, 0, stream>>>(
      ei, den1, ex1);
  edge_aggregate<HEADS, HID><<<blocks((long long)E_TOT * (F1 / 4)), B, 0, stream>>>(
      ei, ex1, h1, aggz);
  elu_bias1<<<blocks((long long)N_NODES * F1), B, 0, stream>>>(
      aggz, bias1, N_NODES * F1);

  // =================== layer 2 ===================
  {
    long long waves = (long long)(N_NODES / 16) * (OUT_CH / 16);
    gemm_xwt_wmma<F1><<<blocks(waves * 32), B, 0, stream>>>(aggz, W2, h2, N_NODES, OUT_CH);
  }
  attn_coef<1, OUT_CH><<<blocks((long long)N_NODES), B, 0, stream>>>(
      h2, att_src2, att_dst2, as2, ad2);
  fill_f32<<<blocks((long long)N_NODES), B, 0, stream>>>(m2, -INFINITY, N_NODES);
  hipMemsetAsync(den2, 0, (size_t)N_NODES * sizeof(float), stream);
  hipMemsetAsync(agg2, 0, (size_t)N_NODES * OUT_CH * sizeof(float), stream);
  edge_lrelu_max<1><<<blocks((long long)E_TOT), B, 0, stream>>>(ei, as2, ad2, m2);
  edge_exp_sum<1><<<blocks((long long)E_TOT), B, 0, stream>>>(ei, as2, ad2, m2, ex2, den2);
  edge_normalize<1><<<blocks((long long)E_TOT), B, 0, stream>>>(ei, den2, ex2);
  edge_aggregate<1, OUT_CH><<<blocks((long long)E_TOT * (OUT_CH / 4)), B, 0, stream>>>(
      ei, ex2, h2, agg2);
  add_bias_out<<<blocks((long long)N_NODES * OUT_CH), B, 0, stream>>>(
      agg2, bias2, out, N_NODES * OUT_CH);
}
